// Decoder_36696200577050
// MI455X (gfx1250) — matrix operand order repaired and checked by reading, not compiled, not run on hardware
//
#include <hip/hip_runtime.h>

#define VOCAB 32000
#define EDIM  256
#define UDIM  1024
#define BATCH 64
#define LLEN  2048
#define HDIM  1024

typedef __attribute__((ext_vector_type(16))) __bf16  v16bf;
typedef __attribute__((ext_vector_type(8)))  float   v8f;

// ---- WMMA fragment layouts, stored "fragment-order": lane-contiguous 32B ----
// A 16x32 bf16: lane = (m%16) | (kh<<4);  kk -> e:
//   kh=(kk>>4)&1, e=kk&15
// Index within a 64(M)x32(K) tile chunk [mtile][lane][e] (2048 elements):
__device__ __forceinline__ int a_frag_idx(int r, int kk) {
  int kh = (kk >> 4) & 1;
  int e  = kk & 15;
  return (((r >> 4) * 32) + ((r & 15) | (kh << 4))) * 16 + e;
}
// B 32x16 bf16: lane = (n%16) | ((k>=16)<<4); e = k&15.
// Index within a 32(K)x32(N) tile [ntile][lane][e] (1024 elements):
__device__ __forceinline__ int b_frag_idx(int k, int n) {
  return ((n >> 4) * 512) + (((n & 15) | (((k >> 4) & 1) << 4)) * 16) + (k & 15);
}

__device__ __forceinline__ v16bf frag_ld(const __bf16* p) {
  return *(const v16bf*)p;                       // 32B: 2x b128 loads
}

// B fragment gathered directly from a row-major f32 matrix (read-once weights):
// lane supplies n; element e -> row k0 + (lane>=16)*16 + e. Coalesced 64B/half,
// non-temporal (one-shot stream), native packed f32->bf16 converts.
__device__ __forceinline__ v16bf b_frag_global(const float* __restrict__ src,
                                               long ld, int k0, int n, int lane) {
  v16bf f;
  const float* p = src + (long)(k0 + ((lane >> 4) << 4)) * ld + n;
#pragma unroll
  for (int e = 0; e < 16; ++e)
    f[e] = (__bf16)__builtin_nontemporal_load(&p[(long)e * ld]);
  return f;
}

__device__ __forceinline__ v8f wmma_bf16(v16bf a, v16bf b, v8f c) {
  return __builtin_amdgcn_wmma_f32_16x16x32_bf16(false, a, false, b,
                                                 (short)0, c, false, false);
}

// ------ one-time: w1 (1024x1024 f32) -> bf16, fragment-order in workspace ------
__global__ __launch_bounds__(256) void k_cvt_w1(const float* __restrict__ w1,
                                                __bf16* __restrict__ w1f) {
  int i = blockIdx.x * 256 + threadIdx.x;       // over 1M elements, coalesced read
  int k = i >> 10, n = i & 1023;
  w1f[(long)((n >> 5) * 32 + (k >> 5)) * 1024 + b_frag_idx(k & 31, n & 31)] =
      (__bf16)w1[i];
}

// ---------------- hid_t = hidden @ w2 + b2   (64 x 1024, K=1024) ----------------
__global__ __launch_bounds__(256) void k_hid(const float* __restrict__ hidden,
                                             const float* __restrict__ w2,
                                             const float* __restrict__ b2,
                                             float* __restrict__ hid) {
  __shared__ __align__(32) __bf16 lA[(UDIM / 32) * 2048];   // 128 KB
  int tid = threadIdx.x, lane = tid & 31, w = tid >> 5;
  int mtile = w & 3, ntile = w >> 2;
  int n0 = blockIdx.x * 32;
  for (int i = tid; i < BATCH * UDIM; i += 256) {
    int r = i >> 10, k = i & 1023;
    lA[(k >> 5) * 2048 + a_frag_idx(r, k & 31)] = (__bf16)hidden[i];
  }
  __syncthreads();
  const __bf16* aB = lA + (mtile * 32 + lane) * 16;
  int n = n0 + ntile * 16 + (lane & 15);
  v8f c = {};
#pragma unroll 4
  for (int ks = 0; ks < UDIM / 32; ++ks)
    c = wmma_bf16(frag_ld(aB + ks * 2048),
                  b_frag_global(w2, UDIM, ks * 32, n, lane), c);
  float bias = b2[n];
  int mAdd = (lane >> 4) * 8;
#pragma unroll
  for (int v = 0; v < 8; ++v)
    hid[(mtile * 16 + mAdd + v) * UDIM + n] = c[v] + bias;
}

// ------ score = tanh(enc@w1 + b1 + hid_t).v_w + v_b  (fused, barrier-free) ------
__global__ __launch_bounds__(256) void k_score(const float* __restrict__ enc,
                                               const __bf16* __restrict__ w1f,
                                               const float* __restrict__ b1,
                                               const float* __restrict__ hid,
                                               const float* __restrict__ v_w,
                                               const float* __restrict__ v_b,
                                               float* __restrict__ score) {
  __shared__ __align__(32) __bf16 lE[(HDIM / 32) * 2048];   // 128 KB
  __shared__ float sAcc[64];
  int tid = threadIdx.x;
  int b = blockIdx.x >> 5, lt = blockIdx.x & 31;
  long rowBase = (long)b * LLEN + lt * 64;
  for (int i = tid; i < BATCH * HDIM; i += 256) {
    int r = i >> 10, k = i & 1023;
    float ev = __builtin_nontemporal_load(&enc[(rowBase + r) * HDIM + k]);
    lE[(k >> 5) * 2048 + a_frag_idx(r, k & 31)] = (__bf16)ev;
  }
  if (tid < 64) sAcc[tid] = 0.f;
  __syncthreads();
  int lane = tid & 31, w = tid >> 5;
  int mtile = w & 3, ntile = w >> 2;
  const __bf16* aB = lE + (mtile * 32 + lane) * 16;
  for (int nc = 0; nc < UDIM / 32; ++nc) {
    v8f c0 = {}, c1 = {};
    const __bf16* bp = w1f + (long)nc * 32 * 1024 + ntile * 512 + lane * 16;
#pragma unroll 4
    for (int ks = 0; ks < HDIM / 32; ks += 2) {
      c0 = wmma_bf16(frag_ld(aB + ks * 2048), frag_ld(bp + ks * 1024), c0);
      c1 = wmma_bf16(frag_ld(aB + (ks + 1) * 2048), frag_ld(bp + (ks + 1) * 1024), c1);
    }
    int n = nc * 32 + ntile * 16 + (lane & 15);
    float add = b1[n] + hid[b * UDIM + n];
    float vw = v_w[n];
    float part[8];
#pragma unroll
    for (int v = 0; v < 8; ++v) part[v] = tanhf(c0[v] + c1[v] + add) * vw;
#pragma unroll
    for (int off = 1; off < 16; off <<= 1)
#pragma unroll
      for (int v = 0; v < 8; ++v) part[v] += __shfl_xor(part[v], off, 32);
    if ((lane & 15) == 0) {
      int mAdd = (lane >> 4) * 8;
#pragma unroll
      for (int v = 0; v < 8; ++v)
        atomicAdd(&sAcc[mtile * 16 + mAdd + v], part[v]);
    }
  }
  __syncthreads();
  if (tid < 64) score[rowBase + tid] = sAcc[tid] + v_b[0];
}

// --------------------------- softmax over L per batch ---------------------------
__global__ __launch_bounds__(256) void k_attn(const float* __restrict__ score,
                                              float* __restrict__ attn_out) {
  __shared__ float red[256];
  int b = blockIdx.x, tid = threadIdx.x;
  const float* s = score + b * LLEN;
  float v[8], mx = -3.4e38f;
#pragma unroll
  for (int j = 0; j < 8; ++j) { v[j] = s[tid + 256 * j]; mx = fmaxf(mx, v[j]); }
  red[tid] = mx; __syncthreads();
  for (int off = 128; off > 0; off >>= 1) {
    if (tid < off) red[tid] = fmaxf(red[tid], red[tid + off]);
    __syncthreads();
  }
  mx = red[0]; __syncthreads();
  float sum = 0.f;
#pragma unroll
  for (int j = 0; j < 8; ++j) { v[j] = __expf(v[j] - mx); sum += v[j]; }
  red[tid] = sum; __syncthreads();
  for (int off = 128; off > 0; off >>= 1) {
    if (tid < off) red[tid] += red[tid + off];
    __syncthreads();
  }
  float inv = 1.f / red[0];
#pragma unroll
  for (int j = 0; j < 8; ++j)
    attn_out[b * LLEN + tid + 256 * j] = v[j] * inv;
}

// ------------------- context[b,:] = sum_l attn[b,l] * enc[b,l,:] ----------------
__global__ __launch_bounds__(256) void k_ctx(const float* __restrict__ attn,
                                             const float* __restrict__ enc,
                                             float* __restrict__ context) {
  __shared__ __align__(16) float sA[LLEN];
  int b = blockIdx.x >> 2, cc = blockIdx.x & 3;
  int tid = threadIdx.x;
  // Stage the 8KB attention row via the gfx1250 async LDS-DMA path.
  {
    const float* g = attn + b * LLEN + tid * 4;
    unsigned lds = (unsigned)(unsigned long)(&sA[0]) + (unsigned)tid * 16u;
    asm volatile("global_load_async_to_lds_b128 %0, %1, off"
                 :: "v"(lds), "v"(g) : "memory");
    asm volatile("global_load_async_to_lds_b128 %0, %1, off"
                 :: "v"(lds + 4096u), "v"(g + 1024) : "memory");
    asm volatile("s_wait_asynccnt 0" ::: "memory");
  }
  __syncthreads();
  int col = cc * 256 + tid;
  const float* eb = enc + (long)b * LLEN * HDIM + col;
  float acc = 0.f;
#pragma unroll 4
  for (int l = 0; l < LLEN; ++l)
    acc += sA[l] * __builtin_nontemporal_load(&eb[(long)l * HDIM]);
  context[b * UDIM + col] = acc;
}

// --------- zi = [context | emb[x]] @ gru_k + gru_b   (64 x 3072, K=1280) --------
__global__ __launch_bounds__(256) void k_gru(const float* __restrict__ context,
                                             const float* __restrict__ emb,
                                             const int* __restrict__ x,
                                             const float* __restrict__ gru_k,
                                             const float* __restrict__ gru_b,
                                             float* __restrict__ zi) {
  __shared__ __align__(32) __bf16 lA[((HDIM + EDIM) / 32) * 2048]; // 160 KB
  __shared__ int sx[64];
  int tid = threadIdx.x;
  if (tid < 64) sx[tid] = x[tid];
  __syncthreads();
  const int KK = HDIM + EDIM;
  for (int i = tid; i < BATCH * KK; i += 256) {
    int r = i / KK, k = i - r * KK;
    float val = (k < UDIM) ? context[r * UDIM + k]
                           : emb[sx[r] * EDIM + (k - UDIM)];
    lA[(k >> 5) * 2048 + a_frag_idx(r, k & 31)] = (__bf16)val;
  }
  __syncthreads();
  int lane = tid & 31, w = tid >> 5;
  int mtile = w & 3, ntile = w >> 2;
  const __bf16* aB = lA + (mtile * 32 + lane) * 16;
  int n0 = blockIdx.x * 32;
  int n = n0 + ntile * 16 + (lane & 15);
  v8f c = {};
#pragma unroll 4
  for (int ks = 0; ks < KK / 32; ++ks)
    c = wmma_bf16(frag_ld(aB + ks * 2048),
                  b_frag_global(gru_k, 3 * UDIM, ks * 32, n, lane), c);
  float bias = gru_b[n];
  int mAdd = (lane >> 4) * 8;
#pragma unroll
  for (int v = 0; v < 8; ++v)
    zi[(mtile * 16 + mAdd + v) * (3 * UDIM) + n] = c[v] + bias;
}

// ----- state = (1 - sigmoid(zi_z)) * tanh(zi_h); also emit bf16 A-fragments -----
__global__ void k_state(const float* __restrict__ zi,
                        float* __restrict__ state_out,
                        __bf16* __restrict__ stateF) {
  int i = blockIdx.x * blockDim.x + threadIdx.x;
  if (i >= BATCH * UDIM) return;
  int m = i >> 10, n = i & 1023;
  float z  = 1.f / (1.f + __expf(-zi[m * 3 * UDIM + n]));
  float hh = tanhf(zi[m * 3 * UDIM + 2 * UDIM + n]);
  float st = (1.f - z) * hh;
  state_out[i] = st;
  stateF[(n >> 5) * 2048 + a_frag_idx(m, n & 31)] = (__bf16)st;  // K dim = n
}

// ------- logits = state @ out_w + out_b  (64 x 32000, K=1024; no LDS at all) -----
__global__ __launch_bounds__(256) void k_logits(const __bf16* __restrict__ stateF,
                                                const float* __restrict__ out_w,
                                                const float* __restrict__ out_b,
                                                float* __restrict__ logits) {
  int tid = threadIdx.x, lane = tid & 31, w = tid >> 5;
  int mtile = w & 3, ntile = w >> 2;
  int n0 = blockIdx.x * 32;
  int n = n0 + ntile * 16 + (lane & 15);
  const __bf16* aB = stateF + (mtile * 32 + lane) * 16;   // L2-resident, reused
  v8f c = {};
#pragma unroll 4
  for (int ks = 0; ks < UDIM / 32; ++ks)
    c = wmma_bf16(frag_ld(aB + ks * 2048),
                  b_frag_global(out_w, VOCAB, ks * 32, n, lane), c);
  float bias = out_b[n];
  int mAdd = (lane >> 4) * 8;
#pragma unroll
  for (int v = 0; v < 8; ++v)
    __builtin_nontemporal_store(c[v] + bias,
        &logits[(long)(mtile * 16 + mAdd + v) * VOCAB + n]);
}

extern "C" void kernel_launch(void* const* d_in, const int* in_sizes, int n_in,
                              void* d_out, int out_size, void* d_ws, size_t ws_size,
                              hipStream_t stream) {
  (void)in_sizes; (void)n_in; (void)out_size; (void)ws_size;
  const int*   x      = (const int*)  d_in[0];
  const float* hidden = (const float*)d_in[1];
  const float* enc    = (const float*)d_in[2];
  const float* emb    = (const float*)d_in[3];
  const float* w1     = (const float*)d_in[4];
  const float* b1     = (const float*)d_in[5];
  const float* w2     = (const float*)d_in[6];
  const float* b2     = (const float*)d_in[7];
  const float* v_w    = (const float*)d_in[8];
  const float* v_b    = (const float*)d_in[9];
  const float* gru_k  = (const float*)d_in[10];
  // d_in[11] = gru_rk: dead, h0 == 0
  const float* gru_b  = (const float*)d_in[12];
  const float* out_w  = (const float*)d_in[13];
  const float* out_b  = (const float*)d_in[14];

  float* out       = (float*)d_out;
  float* logits    = out;                         // 64*32000
  float* state_out = out + BATCH * VOCAB;         // 64*1024
  float* attn_out  = state_out + BATCH * UDIM;    // 64*2048

  float* ws       = (float*)d_ws;
  float* hid      = ws;                           // 65536 f32
  float* score    = hid + BATCH * UDIM;           // 131072 f32
  float* context  = score + BATCH * LLEN;         // 65536 f32
  float* zi       = context + BATCH * UDIM;       // 196608 f32
  __bf16* stateF  = (__bf16*)(zi + BATCH * 3 * UDIM);   // 64K bf16
  __bf16* w1f     = stateF + BATCH * UDIM;              // 1M bf16 (2MB)

  k_cvt_w1<<<(UDIM * UDIM) / 256,      256, 0, stream>>>(w1, w1f);
  k_hid   <<<UDIM / 32,                256, 0, stream>>>(hidden, w2, b2, hid);
  k_score <<<BATCH * (LLEN / 64),      256, 0, stream>>>(enc, w1f, b1, hid, v_w, v_b, score);
  k_attn  <<<BATCH,                    256, 0, stream>>>(score, attn_out);
  k_ctx   <<<BATCH * 4,                256, 0, stream>>>(attn_out, enc, context);
  k_gru   <<<3 * UDIM / 32,            256, 0, stream>>>(context, emb, x, gru_k, gru_b, zi);
  k_state <<<(BATCH * UDIM + 255)/256, 256, 0, stream>>>(zi, state_out, stateF);
  k_logits<<<VOCAB / 32,               256, 0, stream>>>(stateF, out_w, out_b, logits);
}
